// SpectralAttention_44135083933929
// MI455X (gfx1250) — matrix operand order repaired and checked by reading, not compiled, not run on hardware
//
#include <hip/hip_runtime.h>
#include <hip/hip_bf16.h>
#include <math.h>

// ---------------------------------------------------------------------------
// Spectral attention for MI455X (gfx1250), all matmuls via v_wmma_f32_16x16x32_bf16.
// B=4, S=1024, D=1024, H=16, hd=64.
// ---------------------------------------------------------------------------

typedef __bf16 bf16;
typedef __attribute__((ext_vector_type(16))) __bf16 v16bf;
typedef __attribute__((ext_vector_type(8)))  __bf16 v8bf;
typedef __attribute__((ext_vector_type(8)))  float  v8f;

#define DEV __device__ __forceinline__

static constexpr int Bsz = 4;
static constexpr int S   = 1024;
static constexpr int D   = 1024;
static constexpr int H   = 16;
static constexpr int HD  = 64;
static constexpr int BS  = Bsz * S;      // 4096 rows
static constexpr float NEG_BIG = -3.0e38f;

DEV v8f v8f_zero() {
    v8f z;
#pragma unroll
    for (int i = 0; i < 8; ++i) z[i] = 0.f;
    return z;
}

DEV v8bf ld8(const bf16* p) { return *reinterpret_cast<const v8bf*>(p); }

DEV v16bf cat16(v8bf lo, v8bf hi) {
    v16bf r;
#pragma unroll
    for (int i = 0; i < 8; ++i) { r[i] = lo[i]; r[i + 8] = hi[i]; }
    return r;
}

DEV v8f wmma_bf16(v16bf a, v16bf b, v8f c) {
    // D = A(16x32) * B(32x16) + C, fp32 accumulate
    return __builtin_amdgcn_wmma_f32_16x16x32_bf16(
        false, a, false, b, (short)0, c, false, false);
}

DEV float softplus(float x) { return (x > 20.f) ? x : log1pf(__expf(x)); }

// ---------------------------------------------------------------------------
// Prep: fw = reverse cumsum of softplus(freq_raw); pwts = softplus(sym(pool))
// ---------------------------------------------------------------------------
__global__ void spectral_prep_kernel(const float* __restrict__ freq_raw,
                                     const float* __restrict__ pool_raw,
                                     float* __restrict__ fw,
                                     float* __restrict__ pwts) {
    __shared__ float sp[D];
    for (int d = threadIdx.x; d < D; d += blockDim.x)
        sp[d] = softplus(freq_raw[d]);
    __syncthreads();
    if (threadIdx.x == 0) {
        float acc = 0.f;
        for (int d = D - 1; d >= 0; --d) { acc += sp[d]; fw[d] = acc; }
        for (int i = 0; i < 3; ++i)
            pwts[i] = softplus(0.5f * (pool_raw[i] + pool_raw[2 - i]));
    }
}

// ---------------------------------------------------------------------------
// fp32 -> bf16 convert
// ---------------------------------------------------------------------------
__global__ void cvt_bf16_kernel(const float* __restrict__ in,
                                bf16* __restrict__ out, int n) {
    int i = blockIdx.x * blockDim.x + threadIdx.x;
    if (i < n) out[i] = (bf16)in[i];
}

// ---------------------------------------------------------------------------
// WMMA GEMM: C[m,n] = sum_k A[m,k] * W[n,k]   (i.e. A @ W^T)
// Wave tile: 32 (M) x 64 (N) = 8 C tiles; two A row-tiles share four B tiles
// (2x arithmetic intensity vs 16x64). K-loop is software pipelined: next
// k-step's 12 b128 loads are issued before the 8 WMMAs of the current step.
// MODE 0: store bf16, scaled by fw[n]      (Q * fw)
// MODE 1: store fp32                        (K, pre-conv)
// MODE 2: store bf16 transposed [B,H,hd,S]  (V^T for attention B operand)
// MODE 3: store fp32 to d_out               (final O @ Wo^T)
// ---------------------------------------------------------------------------
template <int MODE>
__global__ void wmma_gemm_kernel(const bf16* __restrict__ A,
                                 const bf16* __restrict__ W,
                                 float* __restrict__ outf,
                                 bf16*  __restrict__ outb,
                                 const float* __restrict__ fw) {
    const int wave  = threadIdx.x >> 5;
    const int lane  = threadIdx.x & 31;
    const int ln    = lane & 15;
    const int half  = lane >> 4;
    const int mbase = (blockIdx.x * 8 + wave) * 32;
    const int nbase = blockIdx.y * 64;

    const bf16* ar0  = A + (size_t)(mbase + ln) * D;
    const bf16* ar1  = A + (size_t)(mbase + 16 + ln) * D;
    const bf16* wrow = W + (size_t)(nbase + ln) * D;

    // A operand 16x32: lane=m; e0..7 <-> k=half*8.., e8..15 <-> k=16+half*8..
    auto loadA = [&](const bf16* r, int kb) {
        return cat16(ld8(r + kb + half * 8), ld8(r + kb + 16 + half * 8));
    };
    // B operand 32x16: lane=n; e0..7 <-> k=half*8.., e8..15 <-> k=16+half*8..
    auto loadB = [&](int t, int kb) {
        const bf16* p = wrow + (size_t)t * 16 * D + kb + half * 8;
        return cat16(ld8(p), ld8(p + 16));
    };

    v8f c[8];
#pragma unroll
    for (int t = 0; t < 8; ++t) c[t] = v8f_zero();

    v16bf a0 = loadA(ar0, 0), a1 = loadA(ar1, 0);
    v16bf b0 = loadB(0, 0), b1 = loadB(1, 0), b2 = loadB(2, 0), b3 = loadB(3, 0);

    for (int kb = 32; kb < D; kb += 32) {
        // issue next k-step's loads first (overlaps with WMMAs below)
        v16bf na0 = loadA(ar0, kb), na1 = loadA(ar1, kb);
        v16bf nb0 = loadB(0, kb), nb1 = loadB(1, kb);
        v16bf nb2 = loadB(2, kb), nb3 = loadB(3, kb);

        c[0] = wmma_bf16(a0, b0, c[0]);
        c[1] = wmma_bf16(a0, b1, c[1]);
        c[2] = wmma_bf16(a0, b2, c[2]);
        c[3] = wmma_bf16(a0, b3, c[3]);
        c[4] = wmma_bf16(a1, b0, c[4]);
        c[5] = wmma_bf16(a1, b1, c[5]);
        c[6] = wmma_bf16(a1, b2, c[6]);
        c[7] = wmma_bf16(a1, b3, c[7]);

        a0 = na0; a1 = na1;
        b0 = nb0; b1 = nb1; b2 = nb2; b3 = nb3;
    }
    // drain last k-step
    c[0] = wmma_bf16(a0, b0, c[0]);
    c[1] = wmma_bf16(a0, b1, c[1]);
    c[2] = wmma_bf16(a0, b2, c[2]);
    c[3] = wmma_bf16(a0, b3, c[3]);
    c[4] = wmma_bf16(a1, b0, c[4]);
    c[5] = wmma_bf16(a1, b1, c[5]);
    c[6] = wmma_bf16(a1, b2, c[6]);
    c[7] = wmma_bf16(a1, b3, c[7]);

#pragma unroll
    for (int r = 0; r < 2; ++r) {
#pragma unroll
        for (int t = 0; t < 4; ++t) {
            const int col = nbase + t * 16 + ln;
            float fsc = 1.f;
            if (MODE == 0) fsc = fw[col];
#pragma unroll
            for (int j = 0; j < 8; ++j) {
                const int row = mbase + r * 16 + j + half * 8;  // C: VGPR j -> m=j / m=j+8
                const float v = c[r * 4 + t][j];
                if (MODE == 0) {
                    outb[(size_t)row * D + col] = (bf16)(v * fsc);
                } else if (MODE == 1) {
                    outf[(size_t)row * D + col] = v;
                } else if (MODE == 2) {
                    const int bb = row >> 10, s = row & (S - 1);
                    const int hh = col >> 6,  dd = col & (HD - 1);
                    outb[(((size_t)(bb * H + hh)) * HD + dd) * S + s] = (bf16)v;
                } else {
                    outf[(size_t)row * D + col] = v;
                }
            }
        }
    }
}

// ---------------------------------------------------------------------------
// 3-tap conv over full-D axis of K (pool loop collapsed), fp32 -> bf16
// ---------------------------------------------------------------------------
__global__ void kmix_kernel(const float* __restrict__ Kf,
                            const float* __restrict__ pwts,
                            bf16* __restrict__ Kmix) {
    int i = blockIdx.x * blockDim.x + threadIdx.x;
    if (i >= BS * D) return;
    const int d = i & (D - 1);
    const float p0 = pwts[0], p1 = pwts[1], p2 = pwts[2];
    const float l = (d > 0)     ? Kf[i - 1] : 0.f;
    const float m = Kf[i];
    const float r = (d < D - 1) ? Kf[i + 1] : 0.f;
    Kmix[i] = (bf16)(p0 * l + p1 * m + p2 * r);
}

// ---------------------------------------------------------------------------
// Causal flash attention, one wave per 16 queries of one (b,h).
// Scores computed TRANSPOSED: St = Kmix_tile(A) x Q^T(B)  -> lane = query.
// Two 16-key St tiles pack directly into the K=32 A operand for P@V
// (C layout of St == A-operand layout) with zero cross-lane movement.
// ---------------------------------------------------------------------------
__global__ void spectral_attn_kernel(const bf16* __restrict__ Qf,
                                     const bf16* __restrict__ Kmix,
                                     const bf16* __restrict__ Vt,
                                     bf16* __restrict__ O) {
    const int wid  = (blockIdx.x * blockDim.x + threadIdx.x) >> 5; // 0..4095
    const int lane = threadIdx.x & 31;
    const int ln   = lane & 15;
    const int half = lane >> 4;
    const int qt   = wid & 63;
    const int bh   = wid >> 6;
    const int b    = bh >> 4;
    const int h    = bh & 15;
    const int qbase = qt * 16;
    const int q     = qbase + ln;          // this lane's query (dup in halves)

    // Q^T B operands (d = k dim): lane = query, halves split d range
    const bf16* qp = Qf + ((size_t)(b * S + q)) * D + h * HD;
    const v16bf bq0 = cat16(ld8(qp + half * 8),      ld8(qp + 16 + half * 8));
    const v16bf bq1 = cat16(ld8(qp + 32 + half * 8), ld8(qp + 48 + half * 8));

    float m_run = NEG_BIG, l_run = 0.f;
    v8f acc[4];
#pragma unroll
    for (int t = 0; t < 4; ++t) acc[t] = v8f_zero();

    const int qend = qbase + 15;
    for (int kb = 0; kb <= qend; kb += 32) {
        // --- St tiles: St[key][query] over 2 x 16 keys ---
        v8f st[2];
#pragma unroll
        for (int t = 0; t < 2; ++t) {
            const int key = kb + t * 16 + ln;
            const bf16* kp = Kmix + ((size_t)(b * S + key)) * D + h * HD;
            v16bf a0 = cat16(ld8(kp + half * 8),      ld8(kp + 16 + half * 8));
            v16bf a1 = cat16(ld8(kp + 32 + half * 8), ld8(kp + 48 + half * 8));
            v8f z = v8f_zero();
            z = wmma_bf16(a0, bq0, z);
            z = wmma_bf16(a1, bq1, z);
            st[t] = z;
        }

        // --- scale + causal mask + running max ---
        float mloc = NEG_BIG;
#pragma unroll
        for (int t = 0; t < 2; ++t) {
#pragma unroll
            for (int j = 0; j < 8; ++j) {
                const int key = kb + t * 16 + j + half * 8;
                const float v = (key <= q) ? st[t][j] * 0.125f : NEG_BIG;
                st[t][j] = v;
                mloc = fmaxf(mloc, v);
            }
        }
        mloc = fmaxf(mloc, __shfl_xor(mloc, 16));
        const float m_new = fmaxf(m_run, mloc);
        const float alpha = __expf(m_run - m_new);

        // --- exponentiate; pack P directly as the PV A operand (K=32) ---
        float sloc = 0.f;
        v16bf pa;
#pragma unroll
        for (int j = 0; j < 8; ++j) {
            const float p0v = __expf(st[0][j] - m_new);
            const float p1v = __expf(st[1][j] - m_new);
            sloc += p0v + p1v;
            pa[j]     = (bf16)p0v;   // keys kb + half*8 + j
            pa[j + 8] = (bf16)p1v;   // keys kb + 16 + half*8 + j
        }
        l_run = l_run * alpha + sloc + __shfl_xor(sloc, 16);
        m_run = m_new;

        // broadcast per-query alpha to the out-tile row layout
        float aj[8];
#pragma unroll
        for (int j = 0; j < 8; ++j) aj[j] = __shfl(alpha, j + half * 8);

        // --- acc = acc*alpha + P @ V  (4 d-tiles of 16) ---
#pragma unroll
        for (int t = 0; t < 4; ++t) {
            const bf16* vp = Vt + (((size_t)(b * H + h)) * HD + t * 16 + ln) * S
                           + kb + half * 8;
            v16bf bv = cat16(ld8(vp), ld8(vp + 16));
#pragma unroll
            for (int j = 0; j < 8; ++j) acc[t][j] *= aj[j];
            acc[t] = wmma_bf16(pa, bv, acc[t]);
        }
    }

    const float linv = 1.f / l_run;
    float lj[8];
#pragma unroll
    for (int j = 0; j < 8; ++j) lj[j] = __shfl(linv, j + half * 8);

#pragma unroll
    for (int t = 0; t < 4; ++t) {
#pragma unroll
        for (int j = 0; j < 8; ++j) {
            const int qj = qbase + j + half * 8;
            O[((size_t)(b * S + qj)) * D + h * HD + t * 16 + ln] =
                (bf16)(acc[t][j] * lj[j]);
        }
    }
}

// ---------------------------------------------------------------------------
// Host launcher
// ---------------------------------------------------------------------------
extern "C" void kernel_launch(void* const* d_in, const int* in_sizes, int n_in,
                              void* d_out, int out_size, void* d_ws, size_t ws_size,
                              hipStream_t stream) {
    const float* x    = (const float*)d_in[0];
    const float* Wq   = (const float*)d_in[1];
    const float* Wk   = (const float*)d_in[2];
    const float* Wv   = (const float*)d_in[3];
    const float* Wo   = (const float*)d_in[4];
    const float* freq = (const float*)d_in[5];
    const float* pool = (const float*)d_in[6];

    char* ws = (char*)d_ws;
    float* fw   = (float*)ws;               // 4 KiB
    float* pwts = (float*)(ws + 4096);      // 16 B
    size_t off = 8192;
    bf16* xbf = (bf16*)(ws + off); off += (size_t)BS * D * 2;   // 8 MiB
    bf16* wqb = (bf16*)(ws + off); off += (size_t)D * D * 2;    // 2 MiB
    bf16* wkb = (bf16*)(ws + off); off += (size_t)D * D * 2;
    bf16* wvb = (bf16*)(ws + off); off += (size_t)D * D * 2;
    bf16* wob = (bf16*)(ws + off); off += (size_t)D * D * 2;
    float* Kf = (float*)(ws + off); off += (size_t)BS * D * 4;  // 16 MiB
    bf16* Qfb  = (bf16*)(ws + off); off += (size_t)BS * D * 2;  // 8 MiB
    bf16* Kmx  = (bf16*)(ws + off); off += (size_t)BS * D * 2;
    bf16* Vtb  = (bf16*)(ws + off); off += (size_t)BS * D * 2;
    bf16* Obf  = (bf16*)(ws + off); off += (size_t)BS * D * 2;  // total ~64 MiB

    spectral_prep_kernel<<<1, 256, 0, stream>>>(freq, pool, fw, pwts);

    const int nBSD = BS * D;                 // 4M
    const int nDD  = D * D;                  // 1M
    cvt_bf16_kernel<<<(nBSD + 255) / 256, 256, 0, stream>>>(x,  xbf, nBSD);
    cvt_bf16_kernel<<<(nDD + 255) / 256, 256, 0, stream>>>(Wq, wqb, nDD);
    cvt_bf16_kernel<<<(nDD + 255) / 256, 256, 0, stream>>>(Wk, wkb, nDD);
    cvt_bf16_kernel<<<(nDD + 255) / 256, 256, 0, stream>>>(Wv, wvb, nDD);
    cvt_bf16_kernel<<<(nDD + 255) / 256, 256, 0, stream>>>(Wo, wob, nDD);

    dim3 gg(BS / 256, D / 64);               // 8 waves x 32 rows, 64 cols
    wmma_gemm_kernel<0><<<gg, 256, 0, stream>>>(xbf, wqb, nullptr, Qfb, fw);
    wmma_gemm_kernel<1><<<gg, 256, 0, stream>>>(xbf, wkb, Kf, nullptr, fw);
    wmma_gemm_kernel<2><<<gg, 256, 0, stream>>>(xbf, wvb, nullptr, Vtb, fw);

    kmix_kernel<<<(nBSD + 255) / 256, 256, 0, stream>>>(Kf, pwts, Kmx);

    // 4096 waves (B*H*S/16), 8 waves per block
    spectral_attn_kernel<<<512, 256, 0, stream>>>(Qfb, Kmx, Vtb, Obf);

    wmma_gemm_kernel<3><<<gg, 256, 0, stream>>>(Obf, wob, (float*)d_out, nullptr, fw);
}
